// Net_36661840838916
// MI455X (gfx1250) — matrix operand order repaired and checked by reading, not compiled, not run on hardware
//
#include <hip/hip_runtime.h>
#include <hip/hip_bf16.h>
#include <math.h>

typedef __attribute__((ext_vector_type(16))) _Float16 v16h;
typedef __attribute__((ext_vector_type(8)))  float    v8f;

#define S_NODES   10000
#define E_NODES   20000
#define K_DIM     128
#define FD_DIM    64
#define SEM_DIM   128
#define B_BATCH   2048
#define NE_STU_E  320000
#define NE_EXER_E 320000
#define NE_KN_E   8192

// ---------- monotone float<->int encoding for atomicMax on floats ----------
__device__ __forceinline__ int encF(float f) {
  int i = __float_as_int(f);
  return i >= 0 ? i : (i ^ 0x7FFFFFFF);
}
__device__ __forceinline__ float decF(int i) {
  i = (i >= 0) ? i : (i ^ 0x7FFFFFFF);
  return __int_as_float(i);
}
#define ENC_NEG_INF ((int)0x807FFFFF)

// convert 8 contiguous f32 (b128-loaded) into av[base..base+7]
__device__ __forceinline__ void cvt8(const float* __restrict__ p, v16h& av, int base) {
  float4 x = *(const float4*)p;
  float4 y = *(const float4*)(p + 4);
  av[base + 0] = (_Float16)x.x; av[base + 1] = (_Float16)x.y;
  av[base + 2] = (_Float16)x.z; av[base + 3] = (_Float16)x.w;
  av[base + 4] = (_Float16)y.x; av[base + 5] = (_Float16)y.y;
  av[base + 6] = (_Float16)y.z; av[base + 7] = (_Float16)y.w;
}

// ---------------------------------------------------------------------------
// WMMA GEMM, K compile-time (KSTEPS*32): C[M,N] = A[M,K] @ B[K,N] (+bias[N]).
// One wave per 16x32 macro-tile (two 16x16 WMMA tiles sharing the A fragment).
// f32 in/out, f16 MACs, f32 accumulate. M mult of 16, N mult of 32.
// A/B fragments use contiguous k = 16*half + e (not ISA table order); D per ISA 7.12.2.
// ---------------------------------------------------------------------------
template <int KSTEPS>
__global__ __launch_bounds__(256) void gemm_wmma(
    const float* __restrict__ A, const float* __restrict__ Bm,
    const float* __restrict__ bias, float* __restrict__ C, int M, int N) {
  constexpr int K = KSTEPS * 32;
  int gw   = (int)((blockIdx.x * blockDim.x + threadIdx.x) >> 5);
  int lane = threadIdx.x & 31;
  int tilesN = N >> 5;                      // 32-wide macro tiles
  int tiles  = (M >> 4) * tilesN;
  if (gw >= tiles) return;                  // wave-uniform: EXEC all-1s for WMMA
  int tm = gw / tilesN, tn = gw % tilesN;
  int r = lane & 15, half = lane >> 4;
  const float* arow = A + (size_t)(tm * 16 + r) * K;
  __builtin_prefetch(arow, 0, 1);           // global_prefetch_b8
  int col0 = tn * 32 + r;
  int col1 = col0 + 16;
  v8f acc0 = {}, acc1 = {};
#pragma unroll
  for (int ks = 0; ks < KSTEPS; ++ks) {
    const int kk = ks * 32;
    v16h av, bv0, bv1;
    // A: lanes0-15 K0..15, lanes16-31 K16..31 (b128 loads)
    cvt8(arow + kk + 16 * half,     av, 0);
    cvt8(arow + kk + 16 * half + 8, av, 8);
    // B: lanes0-15 K0..15, lanes16-31 K16..31; n = lane&15
#pragma unroll
    for (int e = 0; e < 16; ++e) {
      const float* brow = Bm + (size_t)(kk + 16 * half + e) * N;
      bv0[e] = (_Float16)brow[col0];
      bv1[e] = (_Float16)brow[col1];
    }
    acc0 = __builtin_amdgcn_wmma_f32_16x16x32_f16(false, av, false, bv0,
                                                  (short)0, acc0, false, false);
    acc1 = __builtin_amdgcn_wmma_f32_16x16x32_f16(false, av, false, bv1,
                                                  (short)0, acc1, false, false);
  }
  float bb0 = bias ? bias[col0] : 0.0f;
  float bb1 = bias ? bias[col1] : 0.0f;
#pragma unroll
  for (int v = 0; v < 8; ++v) {             // D: VGPR v -> row v + 8*half; col = lane&15
    size_t row = (size_t)(tm * 16 + v + 8 * half) * N;
    C[row + col0] = acc0[v] + bb0;
    C[row + col1] = acc1[v] + bb1;
  }
}

// ---------------------------- small utility kernels ------------------------
__global__ void fill_f32_k(float* p, float v, int n) {
  int i = blockIdx.x * blockDim.x + threadIdx.x; if (i < n) p[i] = v;
}
__global__ void fill_i32_k(int* p, int v, int n) {
  int i = blockIdx.x * blockDim.x + threadIdx.x; if (i < n) p[i] = v;
}

// el[n,h] = sum_d z[n,h,d]*al[h,d]; er likewise
__global__ void eler_k(const float* __restrict__ z, const float* __restrict__ al,
                       const float* __restrict__ ar, float* __restrict__ el,
                       float* __restrict__ er, int N) {
  int idx = blockIdx.x * blockDim.x + threadIdx.x;
  if (idx >= N * 8) return;
  int n = idx >> 3, h = idx & 7;
  const float* zp = z + (size_t)n * 64 + h * 8;
  float a = 0.f, b = 0.f;
#pragma unroll
  for (int d = 0; d < 8; ++d) { float zv = zp[d]; a += zv * al[h * 8 + d]; b += zv * ar[h * 8 + d]; }
  el[idx] = a; er[idx] = b;
}

// e = leaky_relu(el[src]+er[dst], 0.2); segment max over dst (encoded atomicMax)
__global__ void edge_max_k(const int* __restrict__ src, const int* __restrict__ dst,
                           const float* __restrict__ el, const float* __restrict__ er,
                           float* __restrict__ e, int* __restrict__ menc, int E) {
  int idx = blockIdx.x * blockDim.x + threadIdx.x;
  if (idx >= E * 8) return;
  int ed = idx >> 3, h = idx & 7;
  int s = src[ed], d = dst[ed];
  float v = el[s * 8 + h] + er[d * 8 + h];
  v = v > 0.f ? v : 0.2f * v;
  e[idx] = v;
  atomicMax(&menc[d * 8 + h], encF(v));
}

// ex = exp(e - m[dst]); segment sum over dst
__global__ void edge_exp_k(const int* __restrict__ dst, float* __restrict__ e,
                           const int* __restrict__ menc, float* __restrict__ ssum, int E) {
  int idx = blockIdx.x * blockDim.x + threadIdx.x;
  if (idx >= E * 8) return;
  int ed = idx >> 3, h = idx & 7;
  int d = dst[ed];
  float ex = expf(e[idx] - decF(menc[d * 8 + h]));
  e[idx] = ex;
  atomicAdd(&ssum[d * 8 + h], ex);
}

// out[dst,h,:] += alpha * z[src,h,:]
__global__ void edge_acc_k(const int* __restrict__ src, const int* __restrict__ dst,
                           const float* __restrict__ e, const float* __restrict__ ssum,
                           const float* __restrict__ z, float* __restrict__ out, int E) {
  int idx = blockIdx.x * blockDim.x + threadIdx.x;
  if (idx >= E * 8) return;
  int ed = idx >> 3, h = idx & 7;
  int s = src[ed], d = dst[ed];
  float alpha = e[idx] / (ssum[d * 8 + h] + 1e-9f);
  const float* zp = z + (size_t)s * 64 + h * 8;
  float* op = out + (size_t)d * 64 + h * 8;
#pragma unroll
  for (int dd = 0; dd < 8; ++dd) atomicAdd(&op[dd], alpha * zp[dd]);
}

__global__ void elu_k(float* x, int n) {
  int i = blockIdx.x * blockDim.x + threadIdx.x;
  if (i < n) { float v = x[i]; x[i] = v > 0.f ? v : (expf(v) - 1.f); }
}

// w_p += (1/N) * sum_s tanh(hidden[n,s]) * semq[s]
__global__ void sem_reduce_k(const float* __restrict__ hidden, const float* __restrict__ sq,
                             float* wdst, float invN, int N) {
  int n = blockIdx.x * blockDim.x + threadIdx.x;
  if (n >= N) return;
  float acc = 0.f;
  for (int s = 0; s < SEM_DIM; ++s) acc += tanhf(hidden[(size_t)n * SEM_DIM + s]) * sq[s];
  atomicAdd(wdst, acc * invN);
}

// fused = softmax(w)[0]*z0 + softmax(w)[1]*z1
__global__ void fuse2_k(const float* __restrict__ z0, const float* __restrict__ z1,
                        const float* __restrict__ w, float* __restrict__ fused, int n) {
  int i = blockIdx.x * blockDim.x + threadIdx.x;
  if (i >= n) return;
  float w0 = w[0], w1 = w[1];
  float m = fmaxf(w0, w1);
  float e0 = expf(w0 - m), e1 = expf(w1 - m);
  float inv = 1.f / (e0 + e1);
  fused[i] = (e0 * z0[i] + e1 * z1[i]) * inv;
}

__global__ void gather_k(const float* __restrict__ tab, const int* __restrict__ ids,
                         float* __restrict__ out, int B, int Kd) {
  int i = blockIdx.x * blockDim.x + threadIdx.x;
  if (i >= B * Kd) return;
  int b = i / Kd, k = i - b * Kd;
  out[i] = tab[(size_t)ids[b] * Kd + k];
}

// Fused prediction head: never materializes [B,K,K] tensors.
// out[b] = sum_i kn_r[b,i]*sig(sum_j (sig(A1[b,j]+C1[i,j])-sig(A2[b,j]+C2[i,j]))*W3[j]+b3)
//          / sum_i kn_r[b,i]
__global__ void predict_k(const float* __restrict__ A1, const float* __restrict__ C1,
                          const float* __restrict__ A2, const float* __restrict__ C2,
                          const float* __restrict__ W3, const float* __restrict__ b3,
                          const float* __restrict__ kn_r, float* __restrict__ out) {
  __shared__ float sv[K_DIM], sw[K_DIM];
  int b = blockIdx.x, i = threadIdx.x;
  const float* a1 = A1 + (size_t)b * K_DIM;
  const float* a2 = A2 + (size_t)b * K_DIM;
  const float* c1 = C1 + (size_t)i * K_DIM;
  const float* c2 = C2 + (size_t)i * K_DIM;
  float acc = 0.f;
  for (int j = 0; j < K_DIM; ++j) {
    float p = 1.f / (1.f + expf(-(a1[j] + c1[j])));
    float q = 1.f / (1.f + expf(-(a2[j] + c2[j])));
    acc += (p - q) * W3[j];
  }
  float o  = 1.f / (1.f + expf(-(acc + b3[0])));
  float kr = kn_r[(size_t)b * K_DIM + i];
  sv[i] = o * kr; sw[i] = kr;
  __syncthreads();
  for (int st = K_DIM / 2; st > 0; st >>= 1) {
    if (i < st) { sv[i] += sv[i + st]; sw[i] += sw[i + st]; }
    __syncthreads();
  }
  if (i == 0) out[b] = sv[0] / sw[0];
}

// ------------------------------- host helpers ------------------------------
static inline int cdiv(int a, int b) { return (a + b - 1) / b; }

static void gemm(const float* A, const float* Bm, const float* bias, float* C,
                 int M, int N, int K, hipStream_t st) {
  int tiles = (M / 16) * (N / 32);
  int thr = tiles * 32;
  if (K == 64)
    gemm_wmma<2><<<cdiv(thr, 256), 256, 0, st>>>(A, Bm, bias, C, M, N);
  else
    gemm_wmma<4><<<cdiv(thr, 256), 256, 0, st>>>(A, Bm, bias, C, M, N);
}

struct Metapath { const int* src; const int* dst; int E;
                  const float* W; const float* al; const float* ar; };

static void run_han(const float* x, int N, int P, const Metapath* mp,
                    const float* sW, const float* sb, const float* sq,
                    const float* pW, const float* pb,
                    float* const* zs, float* z_tmp, float* el, float* er, float* ebuf,
                    int* menc, float* ssum, float* hidden, float* wacc, float* fused,
                    float* out, hipStream_t st) {
  fill_f32_k<<<1, 32, 0, st>>>(wacc, 0.f, 2);
  for (int p = 0; p < P; ++p) {
    gemm(x, mp[p].W, nullptr, z_tmp, N, FD_DIM, K_DIM, st);               // z = x@W
    eler_k<<<cdiv(N * 8, 256), 256, 0, st>>>(z_tmp, mp[p].al, mp[p].ar, el, er, N);
    fill_i32_k<<<cdiv(N * 8, 256), 256, 0, st>>>(menc, ENC_NEG_INF, N * 8);
    fill_f32_k<<<cdiv(N * 8, 256), 256, 0, st>>>(ssum, 0.f, N * 8);
    fill_f32_k<<<cdiv(N * 64, 256), 256, 0, st>>>(zs[p], 0.f, N * 64);
    int E8 = mp[p].E * 8;
    edge_max_k<<<cdiv(E8, 256), 256, 0, st>>>(mp[p].src, mp[p].dst, el, er, ebuf, menc, mp[p].E);
    edge_exp_k<<<cdiv(E8, 256), 256, 0, st>>>(mp[p].dst, ebuf, menc, ssum, mp[p].E);
    edge_acc_k<<<cdiv(E8, 256), 256, 0, st>>>(mp[p].src, mp[p].dst, ebuf, ssum, z_tmp, zs[p], mp[p].E);
    elu_k<<<cdiv(N * 64, 256), 256, 0, st>>>(zs[p], N * 64);
    gemm(zs[p], sW, sb, hidden, N, SEM_DIM, FD_DIM, st);                  // tanh arg
    sem_reduce_k<<<cdiv(N, 256), 256, 0, st>>>(hidden, sq, wacc + p, 1.0f / N, N);
  }
  const float* fsrc = zs[0];
  if (P == 2) {
    fuse2_k<<<cdiv(N * 64, 256), 256, 0, st>>>(zs[0], zs[1], wacc, fused, N * 64);
    fsrc = fused;
  }
  gemm(fsrc, pW, pb, out, N, K_DIM, FD_DIM, st);                          // projection
}

// ------------------------------- entry point -------------------------------
extern "C" void kernel_launch(void* const* d_in, const int* in_sizes, int n_in,
                              void* d_out, int out_size, void* d_ws, size_t ws_size,
                              hipStream_t stream) {
  (void)in_sizes; (void)n_in; (void)out_size; (void)ws_size;
  const float* stu_t  = (const float*)d_in[0];
  const float* exer_t = (const float*)d_in[1];
  const float* kn_t   = (const float*)d_in[2];
  const float* f1W0 = (const float*)d_in[3];  const float* f1al0 = (const float*)d_in[4];
  const float* f1ar0 = (const float*)d_in[5];
  const float* f1sW = (const float*)d_in[6];  const float* f1sb = (const float*)d_in[7];
  const float* f1sq = (const float*)d_in[8];
  const float* f1pW = (const float*)d_in[9];  const float* f1pb = (const float*)d_in[10];
  const float* f3W0 = (const float*)d_in[11]; const float* f3al0 = (const float*)d_in[12];
  const float* f3ar0 = (const float*)d_in[13];
  const float* f3W1 = (const float*)d_in[14]; const float* f3al1 = (const float*)d_in[15];
  const float* f3ar1 = (const float*)d_in[16];
  const float* f3sW = (const float*)d_in[17]; const float* f3sb = (const float*)d_in[18];
  const float* f3sq = (const float*)d_in[19];
  const float* f3pW = (const float*)d_in[20]; const float* f3pb = (const float*)d_in[21];
  const float* f5W0 = (const float*)d_in[22]; const float* f5al0 = (const float*)d_in[23];
  const float* f5ar0 = (const float*)d_in[24];
  const float* f5sW = (const float*)d_in[25]; const float* f5sb = (const float*)d_in[26];
  const float* f5sq = (const float*)d_in[27];
  const float* f5pW = (const float*)d_in[28]; const float* f5pb = (const float*)d_in[29];
  const float* W1 = (const float*)d_in[30];
  const float* W2 = (const float*)d_in[31];
  const float* W3 = (const float*)d_in[32];
  const float* b3 = (const float*)d_in[33];
  const float* kn_r = (const float*)d_in[34];
  const int* stu_id  = (const int*)d_in[35];
  const int* exer_id = (const int*)d_in[36];
  const int* ss0 = (const int*)d_in[37]; const int* sd0 = (const int*)d_in[38];
  const int* es0 = (const int*)d_in[39]; const int* ed0 = (const int*)d_in[40];
  const int* es1 = (const int*)d_in[41]; const int* ed1 = (const int*)d_in[42];
  const int* ks0 = (const int*)d_in[43]; const int* kd0 = (const int*)d_in[44];

  // ---- bump-allocate workspace (~56 MB) ----
  char* wp = (char*)d_ws;
  auto alloc = [&](size_t bytes) -> void* {
    void* r = (void*)wp; wp += (bytes + 255) & ~(size_t)255; return r;
  };
  float* zs_stu  = (float*)alloc((size_t)S_NODES * 64 * 4);
  float* zs_ex0  = (float*)alloc((size_t)E_NODES * 64 * 4);
  float* zs_ex1  = (float*)alloc((size_t)E_NODES * 64 * 4);
  float* zs_kn   = (float*)alloc((size_t)K_DIM * 64 * 4);
  float* han_stu = (float*)alloc((size_t)S_NODES * K_DIM * 4);
  float* han_ex  = (float*)alloc((size_t)E_NODES * K_DIM * 4);
  float* han_kn  = (float*)alloc((size_t)K_DIM * K_DIM * 4);
  float* z_tmp   = (float*)alloc((size_t)E_NODES * 64 * 4);
  float* el      = (float*)alloc((size_t)E_NODES * 8 * 4);
  float* er      = (float*)alloc((size_t)E_NODES * 8 * 4);
  float* ebuf    = (float*)alloc((size_t)NE_EXER_E * 8 * 4);  // aliased as hidden
  float* hidden  = ebuf;  // disjoint lifetimes (stream-ordered)
  int*   menc    = (int*)  alloc((size_t)E_NODES * 8 * 4);
  float* ssum    = (float*)alloc((size_t)E_NODES * 8 * 4);
  float* wacc    = (float*)alloc(256);
  float* fused   = (float*)alloc((size_t)E_NODES * 64 * 4);
  float* bs      = (float*)alloc((size_t)B_BATCH * K_DIM * 4);
  float* be      = (float*)alloc((size_t)B_BATCH * K_DIM * 4);
  float* A1      = (float*)alloc((size_t)B_BATCH * K_DIM * 4);
  float* A2      = (float*)alloc((size_t)B_BATCH * K_DIM * 4);
  float* C1      = (float*)alloc((size_t)K_DIM * K_DIM * 4);
  float* C2      = (float*)alloc((size_t)K_DIM * K_DIM * 4);

  // ---- HAN for knowledge concepts (P=1) ----
  Metapath mkn[1] = {{ks0, kd0, NE_KN_E, f5W0, f5al0, f5ar0}};
  float* zskn[1] = {zs_kn};
  run_han(kn_t, K_DIM, 1, mkn, f5sW, f5sb, f5sq, f5pW, f5pb,
          zskn, z_tmp, el, er, ebuf, menc, ssum, hidden, wacc, fused, han_kn, stream);

  // ---- HAN for exercises (P=2) ----
  Metapath mex[2] = {{es0, ed0, NE_EXER_E, f3W0, f3al0, f3ar0},
                     {es1, ed1, NE_EXER_E, f3W1, f3al1, f3ar1}};
  float* zsex[2] = {zs_ex0, zs_ex1};
  run_han(exer_t, E_NODES, 2, mex, f3sW, f3sb, f3sq, f3pW, f3pb,
          zsex, z_tmp, el, er, ebuf, menc, ssum, hidden, wacc, fused, han_ex, stream);

  // ---- HAN for students (P=1) ----
  Metapath mst[1] = {{ss0, sd0, NE_STU_E, f1W0, f1al0, f1ar0}};
  float* zsst[1] = {zs_stu};
  run_han(stu_t, S_NODES, 1, mst, f1sW, f1sb, f1sq, f1pW, f1pb,
          zsst, z_tmp, el, er, ebuf, menc, ssum, hidden, wacc, fused, han_stu, stream);

  // ---- prediction head ----
  gather_k<<<cdiv(B_BATCH * K_DIM, 256), 256, 0, stream>>>(han_stu, stu_id, bs, B_BATCH, K_DIM);
  gather_k<<<cdiv(B_BATCH * K_DIM, 256), 256, 0, stream>>>(han_ex, exer_id, be, B_BATCH, K_DIM);
  gemm(bs,     W1,                   nullptr, A1, B_BATCH, K_DIM, K_DIM, stream); // bs@W1[:K]
  gemm(han_kn, W1 + K_DIM * K_DIM,   nullptr, C1, K_DIM,   K_DIM, K_DIM, stream); // kn1@W1[K:]
  gemm(be,     W2,                   nullptr, A2, B_BATCH, K_DIM, K_DIM, stream); // be@W2[:K]
  gemm(han_kn, W2 + K_DIM * K_DIM,   nullptr, C2, K_DIM,   K_DIM, K_DIM, stream); // kn1@W2[K:]
  predict_k<<<B_BATCH, K_DIM, 0, stream>>>(A1, C1, A2, C2, W3, b3, kn_r, (float*)d_out);
}
